// WindowAttention_30683246362950
// MI455X (gfx1250) — matrix operand order repaired and checked by reading, not compiled, not run on hardware
//
#include <hip/hip_runtime.h>

typedef unsigned short u16;
typedef unsigned int   u32;

#define DIMC   192
#define NHEAD  6
#define HD     32
#define WSZ    7
#define SHIFTV 3
#define HWDIM  224
#define NWIN   32          // 224/7 windows per side
#define TOK    49
#define TPAD   64

typedef __attribute__((ext_vector_type(16))) __bf16 v16bf;
typedef __attribute__((ext_vector_type(8)))  float  v8f;

struct U32x8 { uint4 a, b; };

__device__ __forceinline__ u16 f2bf(float f) {
  return __builtin_bit_cast(u16, (__bf16)f);   // native v_cvt_*bf16_f32
}

// A-type 16x32 bf16 fragment (ISA 7.12.2 A table):
// row = lane&15; lanes 0-15: K {0..7, 16..23}; lanes 16-31: K {8..15, 24..31}.
__device__ __forceinline__ v16bf load_a(const u16* base, int row0, int k0,
                                        int stride, int lane) {
  const int r    = lane & 15;
  const int half = lane >> 4;
  const u16* p = base + (size_t)(row0 + r) * stride + k0 + half * 8;
  U32x8 t;
  t.a = *(const uint4*)(p);        // K = half*8 .. +7
  t.b = *(const uint4*)(p + 16);   // K = 16 + half*8 .. +7
  return __builtin_bit_cast(v16bf, t);
}

// B-type 32x16 bf16 fragment, source stored row-major [n][k] (i.e. B^T):
// col = lane&15; lanes 0-15: K {0..7, 16..23}; lanes 16-31: K {8..15, 24..31}.
__device__ __forceinline__ v16bf load_b(const u16* base, int col0, int k0,
                                        int stride, int lane) {
  const int c    = lane & 15;
  const int half = lane >> 4;
  const u16* p = base + (size_t)(col0 + c) * stride + k0 + half * 8;
  U32x8 t;
  t.a = *(const uint4*)(p);        // K = half*8 .. +7
  t.b = *(const uint4*)(p + 16);   // K = 16 + half*8 .. +7
  return __builtin_bit_cast(v16bf, t);
}

__device__ __forceinline__ v8f wmma_bf16(v16bf a, v16bf b, v8f c) {
  return __builtin_amdgcn_wmma_f32_16x16x32_bf16(false, a, false, b,
                                                 (short)0, c, false, false);
}

// ---------------------------------------------------------------------------
// Kernel 1: fp32 -> bf16 weight conversion (reads weights from HBM once)
// ---------------------------------------------------------------------------
__global__ void cvt_weights(const float* __restrict__ qkv_w,
                            const float* __restrict__ proj_w,
                            u16* __restrict__ wq, u16* __restrict__ wp) {
  int i = blockIdx.x * 256 + threadIdx.x;
  if (i < 3 * DIMC * DIMC) wq[i] = f2bf(qkv_w[i]);
  if (i < DIMC * DIMC)     wp[i] = f2bf(proj_w[i]);
}

// ---------------------------------------------------------------------------
// Kernel 2: fully fused shifted-window attention. 1 workgroup = 1 window.
// 256 threads = 8 wave32.  LDS = 84 KB (<320 KB/WGP -> multiple WGs resident).
// ---------------------------------------------------------------------------
__global__ __launch_bounds__(256)
void swin_attn(const float* __restrict__ x,
               const u16*  __restrict__ wqkv,   // [576][192] bf16 (L2 resident)
               const float* __restrict__ bqkv,  // [576]
               const u16*  __restrict__ wproj,  // [192][192] bf16
               const float* __restrict__ bproj, // [192]
               float* __restrict__ out) {
  __shared__ __attribute__((aligned(16))) u16   xw[TPAD * DIMC]; // 24 KB tokens x C
  __shared__ __attribute__((aligned(16))) u16   qh[TPAD * HD];   // 4 KB q (pre-scaled)
  __shared__ __attribute__((aligned(16))) u16   kh[TPAD * HD];   // 4 KB k
  __shared__ __attribute__((aligned(16))) u16   vT[HD * TPAD];   // 4 KB v^T [d][j]
  __shared__ __attribute__((aligned(16))) float Sb[TPAD * TPAD]; // 16 KB logits
  __shared__ __attribute__((aligned(16))) u16   Pb[TPAD * TPAD]; // 8 KB probs
  __shared__ __attribute__((aligned(16))) u16   ao[TPAD * DIMC]; // 24 KB attn out

  const int tid  = threadIdx.x;
  const int wave = tid >> 5;          // wave32
  const int lane = tid & 31;
  const int n    = lane & 15;
  const int half = lane >> 4;

  const int wid = blockIdx.x;
  const int b   = wid / (NWIN * NWIN);
  const int wr  = wid % (NWIN * NWIN);
  const int wh  = wr / NWIN;
  const int ww  = wr % NWIN;

  // ---- load shifted window (roll by -SHIFT) into LDS, cast to bf16 --------
  for (int idx = tid; idx < DIMC * TPAD; idx += 256) {
    const int c = idx >> 6;
    const int m = idx & 63;
    float v = 0.f;
    if (m < TOK) {
      const int wy = m / WSZ, wx = m % WSZ;
      const int h = (wh * WSZ + wy + SHIFTV) % HWDIM;
      const int w = (ww * WSZ + wx + SHIFTV) % HWDIM;
      v = x[((size_t)(b * DIMC + c) * HWDIM + h) * HWDIM + w];
    }
    xw[m * DIMC + c] = f2bf(v);
  }
  __syncthreads();

  const float qscale = 0.17677669529663687f;   // 1/sqrt(HD)

  for (int head = 0; head < NHEAD; ++head) {
    // ---- q/k/v slice GEMM: [64x32] = xw[64x192] @ Wslice^T, K=192 ---------
    for (int s = 0; s < 3; ++s) {              // 0=q 1=k 2=v
      const int mt = wave >> 1;
      const int nt = wave & 1;
      const int nrow0 = s * DIMC + head * HD + nt * 16;  // weight row block
      v8f acc = {};
#pragma unroll
      for (int kk = 0; kk < DIMC; kk += 32) {
        v16bf af = load_a(xw,   mt * 16, kk, DIMC, lane);
        v16bf bf = load_b(wqkv, nrow0,   kk, DIMC, lane);  // straight from L2
        acc = wmma_bf16(af, bf, acc);
      }
      const float bias = bqkv[nrow0 + n];
      if (s == 0) {
#pragma unroll
        for (int i = 0; i < 8; ++i)
          qh[(mt * 16 + half * 8 + i) * HD + nt * 16 + n] =
              f2bf((acc[i] + bias) * qscale);
      } else if (s == 1) {
#pragma unroll
        for (int i = 0; i < 8; ++i)
          kh[(mt * 16 + half * 8 + i) * HD + nt * 16 + n] = f2bf(acc[i] + bias);
      } else {
#pragma unroll
        for (int i = 0; i < 8; ++i)                      // store v transposed
          vT[(nt * 16 + n) * TPAD + (mt * 16 + half * 8 + i)] =
              f2bf(acc[i] + bias);
      }
    }
    __syncthreads();

    // ---- S = q @ k^T : 64x64, K=32 (one WMMA per tile, 2 tiles per wave) --
    for (int s = 0; s < 2; ++s) {
      const int t  = wave + 8 * s;
      const int mt = t >> 2, nt = t & 3;
      v16bf af = load_a(qh, mt * 16, 0, HD, lane);
      v16bf bf = load_b(kh, nt * 16, 0, HD, lane);
      v8f acc = {};
      acc = wmma_bf16(af, bf, acc);
#pragma unroll
      for (int i = 0; i < 8; ++i)
        Sb[(mt * 16 + half * 8 + i) * TPAD + nt * 16 + n] = acc[i];
    }
    __syncthreads();

    // ---- masked row softmax: 4 threads per row + wave32 shfl reductions ---
    {
      const int row = tid >> 2;                // 0..63
      const int q4  = tid & 3;
      const float* sr = &Sb[row * TPAD];
      float mx = -1e30f;
      for (int j = q4; j < TOK; j += 4) mx = fmaxf(mx, sr[j]);
      mx = fmaxf(mx, __shfl_xor(mx, 1, 32));
      mx = fmaxf(mx, __shfl_xor(mx, 2, 32));
      float sum = 0.f;
      for (int j = q4; j < TOK; j += 4) sum += __expf(sr[j] - mx);
      sum += __shfl_xor(sum, 1, 32);
      sum += __shfl_xor(sum, 2, 32);
      const float inv = 1.f / sum;
      u16* pr = &Pb[row * TPAD];
      for (int j = q4; j < TPAD; j += 4)
        pr[j] = (j < TOK) ? f2bf(__expf(sr[j] - mx) * inv) : (u16)0;
    }
    __syncthreads();

    // ---- O = P @ v : 64x32, K=64 (1 tile per wave, 2 WMMA) ----------------
    {
      const int mt = wave >> 1, nt = wave & 1;
      v8f acc = {};
#pragma unroll
      for (int kk = 0; kk < TPAD; kk += 32) {
        v16bf af = load_a(Pb, mt * 16, kk, TPAD, lane);
        v16bf bf = load_b(vT, nt * 16, kk, TPAD, lane);
        acc = wmma_bf16(af, bf, acc);
      }
#pragma unroll
      for (int i = 0; i < 8; ++i)
        ao[(mt * 16 + half * 8 + i) * DIMC + head * HD + nt * 16 + n] =
            f2bf(acc[i]);
    }
    __syncthreads();
  }

  // ---- proj GEMM [64x192] @ [192x192]^T, fused merge + roll-back store ----
  for (int s = 0; s < 6; ++s) {                 // 48 tiles over 8 waves
    const int t  = wave + 8 * s;
    const int mt = t / 12, nt = t % 12;
    v8f acc = {};
#pragma unroll
    for (int kk = 0; kk < DIMC; kk += 32) {
      v16bf af = load_a(ao,    mt * 16, kk, DIMC, lane);
      v16bf bf = load_b(wproj, nt * 16, kk, DIMC, lane);
      acc = wmma_bf16(af, bf, acc);
    }
    const int c = nt * 16 + n;
    const float bias = bproj[c];
#pragma unroll
    for (int i = 0; i < 8; ++i) {
      const int m = mt * 16 + half * 8 + i;
      if (m < TOK) {
        const int wy = m / WSZ, wx = m % WSZ;
        const int h = (wh * WSZ + wy + SHIFTV) % HWDIM;
        const int w = (ww * WSZ + wx + SHIFTV) % HWDIM;
        out[((size_t)(b * DIMC + c) * HWDIM + h) * HWDIM + w] = acc[i] + bias;
      }
    }
  }
}

// ---------------------------------------------------------------------------
extern "C" void kernel_launch(void* const* d_in, const int* in_sizes, int n_in,
                              void* d_out, int out_size, void* d_ws, size_t ws_size,
                              hipStream_t stream) {
  const float* x      = (const float*)d_in[0];
  const float* qkv_w  = (const float*)d_in[1];
  const float* qkv_b  = (const float*)d_in[2];
  const float* proj_w = (const float*)d_in[3];
  const float* proj_b = (const float*)d_in[4];

  u16* wq = (u16*)d_ws;                       // [576*192] bf16
  u16* wp = wq + 3 * DIMC * DIMC;             // [192*192] bf16

  const int nconv = 3 * DIMC * DIMC;          // 110592
  hipLaunchKernelGGL(cvt_weights, dim3((nconv + 255) / 256), dim3(256), 0, stream,
                     qkv_w, proj_w, wq, wp);

  const int nwin = 8 * NWIN * NWIN;           // 8192 windows
  hipLaunchKernelGGL(swin_attn, dim3(nwin), dim3(256), 0, stream,
                     x, wq, qkv_b, wp, proj_b, (float*)d_out);
}
